// Attention3D_6098853560438
// MI455X (gfx1250) — matrix operand order repaired and checked by reading, not compiled, not run on hardware
//
#include <hip/hip_runtime.h>

typedef __attribute__((ext_vector_type(16))) _Float16 v16h;
typedef __attribute__((ext_vector_type(8)))  _Float16 v8h;
typedef __attribute__((ext_vector_type(8)))  float    v8f;

constexpr int B_ = 2, N_ = 1024, C_ = 768, H_ = 12, HD_ = 64, HID_ = 64;
constexpr float SCALE = 0.125f; // 64^-0.5

#define DEVFN static __device__ __forceinline__

// A fragment (16x32 f16): row-major source, leading dim ld, tile at (row0, k0).
// lane<16: row=lane, K 0..15; lane>=16: row=lane-16, K 16..31
DEVFN v16h load_a_frag(const _Float16* base, int row0, int ld, int k0, int lane) {
  int hs = lane >> 4, r = lane & 15;
  const _Float16* p = base + (size_t)(row0 + r) * ld + k0 + hs * 16;
  v8h lo = *(const v8h*)(p);
  v8h hi = *(const v8h*)(p + 8);
  return __builtin_shufflevector(lo, hi, 0,1,2,3,4,5,6,7,8,9,10,11,12,13,14,15);
}

// B fragment (32x16 f16): B[k][col] = Mem[(col0+col)*ldc + k0 + k]
// lane<16: col=lane, K 0..15; lane>=16: col=lane-16, K 16..31 -> one v16h load
DEVFN v16h load_b_frag(const _Float16* colbase, int col0, int ldc, int k0, int lane) {
  int hs = lane >> 4, r = lane & 15;
  return *(const v16h*)(colbase + (size_t)(col0 + r) * ldc + k0 + hs * 16);
}

DEVFN v8f wmma_f16(v16h a, v16h b, v8f c) {
  return __builtin_amdgcn_wmma_f32_16x16x32_f16(false, a, false, b, (short)0, c,
                                                false, false);
}

// 32-bit LDS byte offset of a generic pointer into shared memory (VDST operand
// of global_load_async_to_lds_*)
DEVFN unsigned lds_byteoff(const _Float16* p) {
  return (unsigned)(unsigned long long)
         (__attribute__((address_space(3))) const _Float16*)p;
}

// issue one per-lane 16B async copy global -> LDS (tracked by ASYNCcnt)
DEVFN void async_copy_b128(const _Float16* g, const _Float16* l) {
  unsigned loff = lds_byteoff(l);
  asm volatile("global_load_async_to_lds_b128 %0, %1, off"
               :: "v"(loff), "v"(g) : "memory");
}

// ---------------- conversion ----------------
__global__ void k_f32_to_f16(const float* __restrict__ s, _Float16* __restrict__ d, int n) {
  for (int i = blockIdx.x * blockDim.x + threadIdx.x; i < n; i += gridDim.x * blockDim.x)
    d[i] = (_Float16)s[i];
}

// ---------------- per-batch coord range -> 1/(range+eps) ----------------
__global__ void k_coord_range(const float* __restrict__ coords, float* __restrict__ inv_max) {
  int b = blockIdx.x, t = threadIdx.x;
  float mn[3] = {1e30f, 1e30f, 1e30f}, mx[3] = {-1e30f, -1e30f, -1e30f};
  for (int n = t; n < N_; n += blockDim.x)
    for (int c2 = 0; c2 < 3; ++c2) {
      float v = coords[((size_t)b * N_ + n) * 3 + c2];
      mn[c2] = fminf(mn[c2], v);
      mx[c2] = fmaxf(mx[c2], v);
    }
  __shared__ float smn[3][256], smx[3][256];
  for (int c2 = 0; c2 < 3; ++c2) { smn[c2][t] = mn[c2]; smx[c2][t] = mx[c2]; }
  __syncthreads();
  for (int off = 128; off > 0; off >>= 1) {
    if (t < off)
      for (int c2 = 0; c2 < 3; ++c2) {
        smn[c2][t] = fminf(smn[c2][t], smn[c2][t + off]);
        smx[c2][t] = fmaxf(smx[c2][t], smx[c2][t + off]);
      }
    __syncthreads();
  }
  if (t < 3) inv_max[b * 3 + t] = 1.0f / (smx[t][0] - smn[t][0] + 1e-6f);
}

// ---------------- QKV GEMM (2048 x 2304, K=768), scatter to Q,K,V^T ----------------
__global__ void k_qkv_gemm(const _Float16* __restrict__ xh, const _Float16* __restrict__ wh,
                           _Float16* __restrict__ qh, _Float16* __restrict__ kh,
                           _Float16* __restrict__ vth) {
  int wave = threadIdx.x >> 5, lane = threadIdx.x & 31;
  int tile = blockIdx.x * 8 + wave;
  int row0 = (tile / 144) * 16, col0 = (tile % 144) * 16;
  v8f c = {};
  for (int k0 = 0; k0 < C_; k0 += 32)
    c = wmma_f16(load_a_frag(xh, row0, C_, k0, lane),
                 load_b_frag(wh, col0, C_, k0, lane), c);
  int hs = lane >> 4, cl = lane & 15;
  int np = col0 + cl;
  int t3 = np / C_, cc = np % C_;
  int h = cc / HD_, d = cc % HD_;
  for (int r = 0; r < 8; ++r) {
    int mrow = row0 + r + hs * 8;
    int b = mrow >> 10, n = mrow & (N_ - 1);
    _Float16 v = (_Float16)c[r];
    if (t3 == 0)      qh[((size_t)(b * H_ + h) * N_ + n) * HD_ + d] = v;
    else if (t3 == 1) kh[((size_t)(b * H_ + h) * N_ + n) * HD_ + d] = v;
    else              vth[((size_t)(b * H_ + h) * HD_ + d) * N_ + n] = v;
  }
}

// ---------------- rel-pos bias: fused MLP + WMMA vs mlp_w2^T (12 heads, padded to 16) ----
__global__ void k_rel_bias(const float* __restrict__ coords, const float* __restrict__ inv_max,
                           const float* __restrict__ w1, const float* __restrict__ b1,
                           const float* __restrict__ w2, const float* __restrict__ b2,
                           _Float16* __restrict__ rb) {
  int wave = threadIdx.x >> 5, lane = threadIdx.x & 31;
  int hs = lane >> 4, r = lane & 15;
  int P0 = (blockIdx.x * 8 + wave) * 16;       // 16 consecutive (n,m) pairs
  int b = P0 >> 20;                            // N*N = 2^20
  int rem = P0 & (N_ * N_ - 1);
  int n = rem >> 10, m0 = rem & (N_ - 1);
  int m = m0 + r;                              // this lane's A-fragment row (pair)
  float r0 = (coords[((size_t)b * N_ + n) * 3 + 0] - coords[((size_t)b * N_ + m) * 3 + 0]) * inv_max[b * 3 + 0];
  float r1 = (coords[((size_t)b * N_ + n) * 3 + 1] - coords[((size_t)b * N_ + m) * 3 + 1]) * inv_max[b * 3 + 1];
  float r2 = (coords[((size_t)b * N_ + n) * 3 + 2] - coords[((size_t)b * N_ + m) * 3 + 2]) * inv_max[b * 3 + 2];
  int hcol = r;                                // this lane's C/D column (head)
  float cb = (hcol < H_) ? b2[hcol] : 0.0f;
  v8f c;
  for (int i = 0; i < 8; ++i) c[i] = cb;
  for (int s = 0; s < 2; ++s) {                // K = 64 hidden dims, 2 steps of 32
    v16h a, bm;
    for (int e = 0; e < 16; ++e) {
      int j = 32 * s + hs * 16 + e;
      float hv = b1[j] + r0 * w1[j * 3 + 0] + r1 * w1[j * 3 + 1] + r2 * w1[j * 3 + 2];
      a[e] = (_Float16)fmaxf(hv, 0.0f);        // hidden activation, A-fragment slot
      int k = 32 * s + hs * 16 + e;
      bm[e] = (_Float16)((hcol < H_) ? w2[hcol * HID_ + k] : 0.0f);
    }
    c = wmma_f16(a, bm, c);
  }
  if (hcol < H_) {
    size_t base = ((size_t)(b * H_ + hcol) * N_ + n) * N_ + m0 + hs * 8;
    for (int rr = 0; rr < 8; ++rr) rb[base + rr] = (_Float16)c[rr];
  }
}

// ---------------- flash attention ----------------
// One block = one (b,h) x 128 query rows (8 waves x 16). K / V^T tiles for each
// group of 32 keys are staged cooperatively into LDS with
// GLOBAL_LOAD_ASYNC_TO_LDS_B128 (ASYNCcnt), double-buffered, shared by all 8
// waves -> 8x less K/V traffic than per-wave streaming.
__global__ void k_flash_attn(const _Float16* __restrict__ qh, const _Float16* __restrict__ kh,
                             const _Float16* __restrict__ vth, const _Float16* __restrict__ rb,
                             _Float16* __restrict__ aoh) {
  __shared__ __align__(16) _Float16 kbuf[2][32 * HD_];  // [key][d]
  __shared__ __align__(16) _Float16 vbuf[2][HD_ * 32];  // [d][key]
  __shared__ __align__(16) _Float16 plds[8][16 * 32];   // per-wave P staging
  int tid = threadIdx.x;
  int wave = tid >> 5, lane = tid & 31;
  int hs = lane >> 4, cl = lane & 15;
  int bh = blockIdx.x >> 3;                    // 24 (b,h) pairs
  int b = bh / H_, h = bh % H_;
  int n0 = (blockIdx.x & 7) * 128 + wave * 16;
  const _Float16* qb = qh + ((size_t)(b * H_ + h) * N_) * HD_;
  const _Float16* kb = kh + ((size_t)(b * H_ + h) * N_) * HD_;
  const _Float16* vb = vth + ((size_t)(b * H_ + h) * HD_) * N_;
  const _Float16* rbb = rb + ((size_t)(b * H_ + h) * N_ + n0) * N_;

  v16h aq0 = load_a_frag(qb, n0, HD_, 0, lane);
  v16h aq1 = load_a_frag(qb, n0, HD_, 32, lane);
  float mrow[8], lrow[8];
  v8f acc[4];
  for (int r = 0; r < 8; ++r) { mrow[r] = -1e30f; lrow[r] = 0.0f; }
  for (int f = 0; f < 4; ++f) { v8f z = {}; acc[f] = z; }

  // stage 32-key tile at m0 into buffer sel: K tile is contiguous 4KB, V^T tile
  // is 64 rows x 64B. 256 threads x 16B each => one async instr per wave per tile.
  auto issue_tile = [&](int m0, int sel) {
    async_copy_b128(kb + (size_t)m0 * HD_ + tid * 8, &kbuf[sel][tid * 8]);
    int d = tid >> 2, seg = tid & 3;
    async_copy_b128(vb + (size_t)d * N_ + m0 + seg * 8, &vbuf[sel][d * 32 + seg * 8]);
  };

  issue_tile(0, 0);
  constexpr int NIT = N_ / 32;
  for (int it = 0; it < NIT; ++it) {
    int sel = it & 1;
    if (it + 1 < NIT) {
      issue_tile((it + 1) * 32, sel ^ 1);
      asm volatile("s_wait_asynccnt 2" ::: "memory");  // tile `it` fully in LDS
    } else {
      asm volatile("s_wait_asynccnt 0" ::: "memory");
    }
    __syncthreads();
    int m0 = it * 32;
    const _Float16* kt = kbuf[sel];
    const _Float16* vt = vbuf[sel];

    v8f s0 = {}, s1 = {};
    s0 = wmma_f16(aq0, *(const v16h*)&kt[cl * HD_ + hs * 16], s0);
    s0 = wmma_f16(aq1, *(const v16h*)&kt[cl * HD_ + 32 + hs * 16], s0);
    s1 = wmma_f16(aq0, *(const v16h*)&kt[(16 + cl) * HD_ + hs * 16], s1);
    s1 = wmma_f16(aq1, *(const v16h*)&kt[(16 + cl) * HD_ + 32 + hs * 16], s1);

    float p0[8], p1[8], alpha[8];
    for (int r = 0; r < 8; ++r) {
      int nr = r + hs * 8;
      p0[r] = s0[r] * SCALE + (float)rbb[(size_t)nr * N_ + m0 + cl];
      p1[r] = s1[r] * SCALE + (float)rbb[(size_t)nr * N_ + m0 + 16 + cl];
      float t = fmaxf(p0[r], p1[r]);
      for (int msk = 1; msk < 16; msk <<= 1) t = fmaxf(t, __shfl_xor(t, msk, 32));
      float nm = fmaxf(mrow[r], t);
      alpha[r] = __expf(mrow[r] - nm);
      mrow[r] = nm;
      p0[r] = __expf(p0[r] - nm);
      p1[r] = __expf(p1[r] - nm);
      float rs = p0[r] + p1[r];
      for (int msk = 1; msk < 16; msk <<= 1) rs += __shfl_xor(rs, msk, 32);
      lrow[r] = lrow[r] * alpha[r] + rs;
    }
    // P: C-layout -> A-layout through per-wave LDS (DS ops in-order within wave)
    for (int r = 0; r < 8; ++r) {
      int row = r + hs * 8;
      plds[wave][row * 32 + cl]      = (_Float16)p0[r];
      plds[wave][row * 32 + 16 + cl] = (_Float16)p1[r];
    }
    __builtin_amdgcn_wave_barrier();
    v8h lo = *(const v8h*)&plds[wave][cl * 32 + hs * 16];
    v8h hi = *(const v8h*)&plds[wave][cl * 32 + hs * 16 + 8];
    v16h ap = __builtin_shufflevector(lo, hi, 0,1,2,3,4,5,6,7,8,9,10,11,12,13,14,15);
    for (int f = 0; f < 4; ++f) {
      for (int r = 0; r < 8; ++r) acc[f][r] *= alpha[r];
      acc[f] = wmma_f16(ap, *(const v16h*)&vt[(f * 16 + cl) * 32 + hs * 16], acc[f]);
    }
    __syncthreads();  // tile consumed; safe to overwrite this buffer next round
  }
  for (int f = 0; f < 4; ++f)
    for (int r = 0; r < 8; ++r) {
      int row = r + hs * 8;
      aoh[((size_t)(b * N_ + n0 + row)) * C_ + h * HD_ + f * 16 + cl] =
          (_Float16)(acc[f][r] / lrow[r]);
    }
}

// ---------------- output projection (2048 x 768, K=768) + bias ----------------
__global__ void k_proj_gemm(const _Float16* __restrict__ aoh, const _Float16* __restrict__ pwh,
                            const float* __restrict__ pb, float* __restrict__ out) {
  int wave = threadIdx.x >> 5, lane = threadIdx.x & 31;
  int tile = blockIdx.x * 8 + wave;
  int row0 = (tile / 48) * 16, col0 = (tile % 48) * 16;
  v8f c = {};
  for (int k0 = 0; k0 < C_; k0 += 32)
    c = wmma_f16(load_a_frag(aoh, row0, C_, k0, lane),
                 load_b_frag(pwh, col0, C_, k0, lane), c);
  int hs = lane >> 4, cl = lane & 15;
  float bias = pb[col0 + cl];
  for (int r = 0; r < 8; ++r)
    out[(size_t)(row0 + r + hs * 8) * C_ + col0 + cl] = c[r] + bias;
}

extern "C" void kernel_launch(void* const* d_in, const int* in_sizes, int n_in,
                              void* d_out, int out_size, void* d_ws, size_t ws_size,
                              hipStream_t stream) {
  (void)in_sizes; (void)n_in; (void)out_size; (void)ws_size;
  const float* x      = (const float*)d_in[0];
  const float* coords = (const float*)d_in[1];
  const float* qkv_w  = (const float*)d_in[2];
  const float* proj_w = (const float*)d_in[3];
  const float* proj_b = (const float*)d_in[4];
  const float* mlp_w1 = (const float*)d_in[5];
  const float* mlp_b1 = (const float*)d_in[6];
  const float* mlp_w2 = (const float*)d_in[7];
  const float* mlp_b2 = (const float*)d_in[8];
  float* out = (float*)d_out;

  char* ws = (char*)d_ws;
  size_t off = 0;
  auto alloc = [&](size_t bytes) {
    void* p = ws + off;
    off = (off + bytes + 255) & ~(size_t)255;
    return p;
  };
  _Float16* xh  = (_Float16*)alloc((size_t)B_ * N_ * C_ * 2);
  _Float16* wh  = (_Float16*)alloc((size_t)3 * C_ * C_ * 2);
  _Float16* pwh = (_Float16*)alloc((size_t)C_ * C_ * 2);
  _Float16* qh  = (_Float16*)alloc((size_t)B_ * H_ * N_ * HD_ * 2);
  _Float16* kh  = (_Float16*)alloc((size_t)B_ * H_ * N_ * HD_ * 2);
  _Float16* vth = (_Float16*)alloc((size_t)B_ * H_ * N_ * HD_ * 2);
  _Float16* aoh = (_Float16*)alloc((size_t)B_ * N_ * C_ * 2);
  float*    inv = (float*)alloc(8 * sizeof(float));
  _Float16* rbh = (_Float16*)alloc((size_t)B_ * H_ * N_ * N_ * 2);

  k_f32_to_f16<<<512, 256, 0, stream>>>(x, xh, B_ * N_ * C_);
  k_f32_to_f16<<<512, 256, 0, stream>>>(qkv_w, wh, 3 * C_ * C_);
  k_f32_to_f16<<<512, 256, 0, stream>>>(proj_w, pwh, C_ * C_);
  k_coord_range<<<B_, 256, 0, stream>>>(coords, inv);
  k_qkv_gemm<<<2304, 256, 0, stream>>>(xh, wh, qh, kh, vth);        // 128*144 tiles / 8 waves
  k_rel_bias<<<16384, 256, 0, stream>>>(coords, inv, mlp_w1, mlp_b1, mlp_w2, mlp_b2, rbh);
  k_flash_attn<<<192, 256, 0, stream>>>(qh, kh, vth, rbh, aoh);     // 24 (b,h) x 8 q-blocks
  k_proj_gemm<<<768, 256, 0, stream>>>(aoh, pwh, proj_b, out);      // 128*48 tiles / 8 waves
}
